// CRF_47287589929160
// MI455X (gfx1250) — compile-verified, matrix operand order checked by reading, not run on hardware
//
#include <hip/hip_runtime.h>

typedef __attribute__((ext_vector_type(16))) __bf16 v16bf;
typedef __attribute__((ext_vector_type(8)))  float  v8f;

#define CRF_B 512
#define CRF_T 1024
#define CRF_L 64
#define ROWS_PER_BLK 16
#define FWD_BLOCKS (CRF_B / ROWS_PER_BLK)   // 32 chains of 16 batch rows
#define GOLD_BLOCKS 256
#define PF_DIST 4                            // prefetch distance in time-steps

#define LOG2E 1.4426950408889634f
#define LN2   0.6931471805599453f

// workspace layout (bytes)
#define WS_EXPT_OFF   0        // exp(transition) in WMMA-B layout: 4 tiles * 2 Khalves * 32 lanes * 16 bf16 = 8192 B
#define WS_TSTOP_OFF  8192     // transition[:,STOP] * log2e : 64 f32
#define WS_LOGZ_OFF   8448     // 32 f32 per-block logZ partials
#define WS_GOLD_OFF   8576     // 256 f32 per-block gold partials

__device__ __forceinline__ unsigned short f2bf(float f) {
  unsigned int u = __float_as_uint(f);
  u += 0x7FFFu + ((u >> 16) & 1u);      // round-to-nearest-even
  return (unsigned short)(u >> 16);
}

// WGP-scope prefetch (scope field 0 = pull into all cache levels on miss).
// __builtin_prefetch can only reach SE/SYS scope, which skips the WGP$.
__device__ __forceinline__ void prefetch_wgp(const void* p) {
  asm volatile("global_prefetch_b8 %0, off" :: "v"((unsigned long long)(uintptr_t)p));
}

// ---------------------------------------------------------------------------
// Kernel 0: precompute exp(transition) pre-swizzled into the 16-bit WMMA
// B-operand layout (B is KxN = prev x next; lane = N%16 + 16*Khalf-lane,
// element e <-> K = (e<8 ? e : e+8) + 8*(lane/16) within the 32-K block),
// plus transition[:,STOP] pre-scaled to the log2 domain.
// ---------------------------------------------------------------------------
__global__ __launch_bounds__(256) void crf_prep(const float* __restrict__ transition,
                                                unsigned short* __restrict__ expT,
                                                float* __restrict__ tstop) {
  const int tid   = threadIdx.x;        // 256 threads == 4 tiles * 2 halves * 32 lanes
  const int w     = tid >> 6;           // N-tile (next-tags 16w..16w+15)
  const int h     = (tid >> 5) & 1;     // K-half (prev 32h..32h+31)
  const int lane  = tid & 31;
  const int n     = w * 16 + (lane & 15);
  const int halfL = lane >> 4;
  unsigned short* dst = expT + (size_t)tid * 16;
#pragma unroll
  for (int e = 0; e < 16; ++e) {
    int k    = (e < 8 ? e : e + 8) + 8 * halfL;
    int prev = h * 32 + k;
    // exp(x) = exp2(x * log2e); hardware v_exp_f32 is exp2
    dst[e]   = f2bf(__builtin_amdgcn_exp2f(transition[prev * CRF_L + n] * LOG2E));
  }
  if (tid < CRF_L) tstop[tid] = transition[tid * CRF_L + 2] * LOG2E;   // STOP = 2
}

// ---------------------------------------------------------------------------
// Kernel 1: forward recursion, entirely in the log2 domain.
// beta = alpha * log2e.  One block = 16 batch rows, 4 waves; wave w owns
// next-tag columns [16w,16w+16). Per step:
//   m = rowmax(beta); aExp = bf16(exp2(beta-m));           (bare v_exp_f32)
//   C = aExp @ exp(Tr)  (2x v_wmma_f32_16x16x32_bf16, f32 accumulate);
//   beta' = fma(emit, log2e, m) + log2(C)                  (bare v_log_f32)
// ---------------------------------------------------------------------------
__global__ __launch_bounds__(128) void crf_forward(const float* __restrict__ emit,
                                                   const float* __restrict__ transition,
                                                   const unsigned short* __restrict__ expT,
                                                   const float* __restrict__ tstop,
                                                   float* __restrict__ logZPart) {
  __shared__ float betaF[16][72];                         // padded: 288 B row stride
  __shared__ __align__(16) unsigned short aExp[16][72];   // padded: 144 B row stride (conflict-free b128 loads)
  __shared__ float mRow[16];
  __shared__ float rowZ[16];

  const int tid   = threadIdx.x;
  const int wave  = tid >> 5;
  const int lane  = tid & 31;
  const int row8  = tid >> 3;      // 16 reduction rows, 8 lanes each
  const int sub   = tid & 7;
  const int halfL = lane >> 4;
  const int n     = wave * 16 + (lane & 15);   // this lane's next-tag column
  const size_t b0 = (size_t)blockIdx.x * ROWS_PER_BLK;

  union Frag { uint4 u[2]; v16bf v; };

  // B fragments (exp(transition)) live in registers for the whole chain.
  Frag fB0, fB1;
  {
    const uint4* p0 = (const uint4*)(expT + ((size_t)(wave * 2 + 0) * 32 + lane) * 16);
    fB0.u[0] = p0[0]; fB0.u[1] = p0[1];
    const uint4* p1 = (const uint4*)(expT + ((size_t)(wave * 2 + 1) * 32 + lane) * 16);
    fB1.u[0] = p1[0]; fB1.u[1] = p1[1];
  }

  // beta0 = (emit[:,1,:] + transition[START=1,:]) * log2e
#pragma unroll
  for (int k = 0; k < 8; ++k) {
    int c = sub * 8 + k;
    betaF[row8][c] =
        (emit[((b0 + row8) * CRF_T + 1) * CRF_L + c] + transition[1 * CRF_L + c]) * LOG2E;
  }
  __syncthreads();

  for (int t = 2; t <= CRF_T - 2; ++t) {
    // Issue this step's emit loads early (latency hidden behind the reduction).
    float ev[8];
#pragma unroll
    for (int r = 0; r < 8; ++r) {
      int M = r + 8 * halfL;
      ev[r] = emit[((b0 + M) * CRF_T + t) * CRF_L + n];
    }
    // Prefetch the emit tile PF_DIST steps ahead at WGP scope
    // (one 64B line per (row, wave-tile); 64 lanes cover the 4KB tile).
    if (lane < 16 && t + PF_DIST <= CRF_T - 2)
      prefetch_wgp(&emit[((b0 + lane) * CRF_T + (t + PF_DIST)) * CRF_L + wave * 16]);

    // Per-row max: 8 lanes per row, wave32 butterfly.
    float av[8];
#pragma unroll
    for (int k = 0; k < 8; ++k) av[k] = betaF[row8][sub * 8 + k];
    float mx = av[0];
#pragma unroll
    for (int k = 1; k < 8; ++k) mx = fmaxf(mx, av[k]);
    mx = fmaxf(mx, __shfl_xor(mx, 1, 32));
    mx = fmaxf(mx, __shfl_xor(mx, 2, 32));
    mx = fmaxf(mx, __shfl_xor(mx, 4, 32));
    if (sub == 0) mRow[row8] = mx;

    // Stage 2^(beta - m) as bf16 (bare v_exp_f32 per element).
#pragma unroll
    for (int k = 0; k < 8; ++k)
      aExp[row8][sub * 8 + k] = f2bf(__builtin_amdgcn_exp2f(av[k] - mx));
    __syncthreads();

    // A fragments: each = two 16B LDS loads (K 0..7/16..23 etc. per the 16-bit
    // A layout). 144 B row stride -> the 16 row loads land on disjoint banks.
    const int M0 = lane & 15;
    Frag fA0, fA1;
    fA0.u[0] = *(const uint4*)&aExp[M0][ 0 + 8 * halfL];
    fA0.u[1] = *(const uint4*)&aExp[M0][16 + 8 * halfL];
    fA1.u[0] = *(const uint4*)&aExp[M0][32 + 8 * halfL];
    fA1.u[1] = *(const uint4*)&aExp[M0][48 + 8 * halfL];

    v8f acc = {0.f, 0.f, 0.f, 0.f, 0.f, 0.f, 0.f, 0.f};
    acc = __builtin_amdgcn_wmma_f32_16x16x32_bf16(false, fA0.v, false, fB0.v,
                                                  (short)0, acc, false, false);
    acc = __builtin_amdgcn_wmma_f32_16x16x32_bf16(false, fA1.v, false, fB1.v,
                                                  (short)0, acc, false, false);

    // beta' = fma(emit, log2e, m) + log2(C) (bare v_log_f32; C >= exp(-15) so
    // always normal). D layout: element r -> M = r + 8*(lane/16), N = lane%16.
#pragma unroll
    for (int r = 0; r < 8; ++r) {
      int M = r + 8 * halfL;
      betaF[M][n] = fmaf(ev[r], LOG2E, mRow[M]) + __builtin_amdgcn_logf(acc[r]);
    }
    __syncthreads();
  }

  // Close: logZ_row = ln2 * log2sumexp2(beta + tstop*log2e); deterministic butterflies.
  float xv[8];
  float mx = -3.0e38f;
#pragma unroll
  for (int k = 0; k < 8; ++k) {
    int c = sub * 8 + k;
    xv[k] = betaF[row8][c] + tstop[c];
    mx = fmaxf(mx, xv[k]);
  }
  mx = fmaxf(mx, __shfl_xor(mx, 1, 32));
  mx = fmaxf(mx, __shfl_xor(mx, 2, 32));
  mx = fmaxf(mx, __shfl_xor(mx, 4, 32));
  float s = 0.f;
#pragma unroll
  for (int k = 0; k < 8; ++k) s += __builtin_amdgcn_exp2f(xv[k] - mx);
  s += __shfl_xor(s, 1, 32);
  s += __shfl_xor(s, 2, 32);
  s += __shfl_xor(s, 4, 32);
  if (sub == 0) rowZ[row8] = LN2 * (mx + __builtin_amdgcn_logf(s));
  __syncthreads();
  if (tid == 0) {
    float z = 0.f;
    for (int r = 0; r < 16; ++r) z += rowZ[r];     // fixed order -> deterministic
    logZPart[blockIdx.x] = z;
  }
}

// ---------------------------------------------------------------------------
// Kernel 2: gold score (gather-reduce), deterministic tree reduction per block.
// ---------------------------------------------------------------------------
__global__ __launch_bounds__(256) void crf_gold(const float* __restrict__ emit,
                                                const int* __restrict__ target,
                                                const float* __restrict__ transition,
                                                float* __restrict__ goldPart) {
  __shared__ float red[256];
  const int tid = threadIdx.x;
  float acc = 0.f;
  const long total = (long)CRF_B * (CRF_T - 1);
  for (long i = (long)blockIdx.x * blockDim.x + tid; i < total;
       i += (long)gridDim.x * blockDim.x) {
    int b  = (int)(i / (CRF_T - 1));
    int t  = (int)(i % (CRF_T - 1));
    int g0 = target[(size_t)b * CRF_T + t];
    int g1 = target[(size_t)b * CRF_T + t + 1];
    acc += transition[g0 * CRF_L + g1];
    if (t >= 1) acc += emit[((size_t)b * CRF_T + t) * CRF_L + g0];
  }
  red[tid] = acc;
  __syncthreads();
  for (int s = 128; s > 0; s >>= 1) {
    if (tid < s) red[tid] += red[tid + s];
    __syncthreads();
  }
  if (tid == 0) goldPart[blockIdx.x] = red[0];
}

// ---------------------------------------------------------------------------
// Kernel 3: final fixed-order combine; out = logZ - gold.
// ---------------------------------------------------------------------------
__global__ void crf_final(const float* __restrict__ logZPart,
                          const float* __restrict__ goldPart,
                          float* __restrict__ out) {
  if (threadIdx.x == 0 && blockIdx.x == 0) {
    float z = 0.f;
    for (int i = 0; i < FWD_BLOCKS; ++i) z += logZPart[i];
    float g = 0.f;
    for (int i = 0; i < GOLD_BLOCKS; ++i) g += goldPart[i];
    out[0] = z - g;
  }
}

extern "C" void kernel_launch(void* const* d_in, const int* in_sizes, int n_in,
                              void* d_out, int out_size, void* d_ws, size_t ws_size,
                              hipStream_t stream) {
  const float* emit       = (const float*)d_in[0];
  const int*   target     = (const int*)d_in[1];
  const float* transition = (const float*)d_in[2];
  float* out = (float*)d_out;
  char*  ws  = (char*)d_ws;

  unsigned short* expT    = (unsigned short*)(ws + WS_EXPT_OFF);
  float* tstop            = (float*)(ws + WS_TSTOP_OFF);
  float* logZPart         = (float*)(ws + WS_LOGZ_OFF);
  float* goldPart         = (float*)(ws + WS_GOLD_OFF);

  crf_prep<<<1, 256, 0, stream>>>(transition, expT, tstop);
  crf_forward<<<FWD_BLOCKS, 128, 0, stream>>>(emit, transition, expT, tstop, logZPart);
  crf_gold<<<GOLD_BLOCKS, 256, 0, stream>>>(emit, target, transition, goldPart);
  crf_final<<<1, 1, 0, stream>>>(logZPart, goldPart, out);
}
